// quantLinLayer_83519934038114
// MI455X (gfx1250) — 8-bit matrix operand order repaired and checked by reading, compiled, not run on hardware
//
#include <hip/hip_runtime.h>

typedef int   v8i __attribute__((ext_vector_type(8)));
typedef int   v4i __attribute__((ext_vector_type(4)));
typedef float v4f __attribute__((ext_vector_type(4)));

#define IN_F   1024
#define OUT_F  1024
#define ROWS   (16 * 4096)
#define NGA    16
#define NGW    16
#define NTILES (OUT_F / 16)
#define RTILES (ROWS / 16)

#if __has_builtin(__builtin_amdgcn_sched_group_barrier)
#define SGB(mask, n) __builtin_amdgcn_sched_group_barrier((mask), (n), 0)
#else
#define SGB(mask, n)
#endif

__device__ __forceinline__ unsigned pack4(int a, int b, int c, int d) {
  return (unsigned)(a & 255) | ((unsigned)(b & 255) << 8) |
         ((unsigned)(c & 255) << 16) | ((unsigned)(d & 255) << 24);
}

__device__ __forceinline__ int quant_act(float v, float inv, float qmin, float qmax) {
  float q = __builtin_rintf(v * inv);
  q = __builtin_fminf(q, qmax);
  q = __builtin_fmaxf(q, qmin);
  return (int)q;
}

__device__ __forceinline__ v8i loadB(const signed char* bp, int g) {
  v4i b0 = *(const v4i*)(bp + g * 1024);
  v4i b1 = *(const v4i*)(bp + g * 1024 + 512);
  v8i b;
  b[0] = b0[0]; b[1] = b0[1]; b[2] = b0[2]; b[3] = b0[3];
  b[4] = b1[0]; b[5] = b1[1]; b[6] = b1[2]; b[7] = b1[3];
  return b;
}

__device__ __forceinline__ bool alpha_uniform(const float* __restrict__ alpha) {
  bool uni = true;
#pragma unroll
  for (int g = 1; g < NGA; ++g) uni = uni && (alpha[g] == alpha[0]);
  return uni;
}

__device__ __forceinline__ void build_A(const float* __restrict__ xr, int hi,
                                        const float* __restrict__ alpha,
                                        const int* __restrict__ actbits,
                                        v8i* aReg, float* scaleA) {
#pragma unroll
  for (int g = 0; g < NGA; ++g) {
    float al  = alpha[g];
    scaleA[g] = __builtin_exp2f(al);
    float inv = __builtin_exp2f(-al);
    float p   = __builtin_exp2f((float)(actbits[g] - 1));
    float qmaxf = p - 1.0f, qminf = -p;
#pragma unroll
    for (int c = 0; c < 4; ++c) {
      int k0 = g * 64 + (c >> 1) * 32 + hi * 16 + (c & 1) * 8;
      v4f f0 = *(const v4f*)(xr + k0);
      v4f f1 = *(const v4f*)(xr + k0 + 4);
      aReg[g][2 * c] = (int)pack4(
          quant_act(f0[0], inv, qminf, qmaxf), quant_act(f0[1], inv, qminf, qmaxf),
          quant_act(f0[2], inv, qminf, qmaxf), quant_act(f0[3], inv, qminf, qmaxf));
      aReg[g][2 * c + 1] = (int)pack4(
          quant_act(f1[0], inv, qminf, qmaxf), quant_act(f1[1], inv, qminf, qmaxf),
          quant_act(f1[2], inv, qminf, qmaxf), quant_act(f1[3], inv, qminf, qmaxf));
    }
  }
}

__global__ __launch_bounds__(256) void wq_swizzle_kernel(
    const float* __restrict__ w, const float* __restrict__ phi,
    const float* __restrict__ delta, const int* __restrict__ bit,
    signed char* __restrict__ wsB) {
  int id   = blockIdx.x * 256 + threadIdx.x;
  int lane = id & 31;
  int half = (id >> 5) & 1;
  int g    = (id >> 6) & 15;
  int nt   = id >> 10;
  int o    = nt * 16 + (lane & 15);
  int hi   = (lane >> 4) & 1;
  int kb   = g * 64 + half * 32 + hi * 16;

  int   og    = o >> 6;
  float inv   = __builtin_exp2f(-delta[og]);
  float ph    = phi[og];
  float qmaxf = __builtin_exp2f((float)(bit[og] - 1)) - 1.0f;

  const float* src = w + (size_t)o * IN_F + kb;
  unsigned pk[4];
#pragma unroll
  for (int d4 = 0; d4 < 4; ++d4) {
    v4f f = *(const v4f*)(src + d4 * 4);
    int q[4];
#pragma unroll
    for (int j = 0; j < 4; ++j) {
      float a   = f[j];
      float idx = __builtin_floorf(__builtin_fabsf(a) * inv + ph);
      idx       = __builtin_fminf(idx, qmaxf);
      idx       = __builtin_fmaxf(idx, 0.0f);
      int qi    = (int)idx;
      q[j]      = (a < 0.0f) ? -qi : qi;
    }
    pk[d4] = pack4(q[0], q[1], q[2], q[3]);
  }
  v4i st = {(int)pk[0], (int)pk[1], (int)pk[2], (int)pk[3]};
  *(v4i*)(wsB + (size_t)id * 16) = st;
}

__global__ __launch_bounds__(256) void gemm_iu8_uniform_kernel(
    const float* __restrict__ x, const float* __restrict__ bias,
    const float* __restrict__ delta, const float* __restrict__ alpha,
    const int* __restrict__ actbits, const signed char* __restrict__ wsB,
    float* __restrict__ out) {
  if (!alpha_uniform(alpha)) return;

  int lane = threadIdx.x & 31;
  int wave = threadIdx.x >> 5;
  int tile = blockIdx.x * 8 + wave;
  int m    = lane & 15;
  int hi   = lane >> 4;
  int row0 = tile * 16;
  const float* xr = x + (size_t)(row0 + m) * IN_F;

  float scaleA[NGA];
  v8i   aReg[NGA];
  build_A(xr, hi, alpha, actbits, aReg, scaleA);
  float s0 = scaleA[0];

  v8i    dPrev  = {0, 0, 0, 0, 0, 0, 0, 0};
  float  sPrev  = 0.0f, bsPrev = 0.0f;
  float* opPrev = out + (size_t)(row0 + hi * 8) * OUT_F + m;

#pragma unroll 1
  for (int nt = 0; nt < NTILES; ++nt) {
    int   col  = nt * 16 + m;
    float sAll = s0 * __builtin_exp2f(delta[nt >> 2]);
    float bs   = bias[col];
    const signed char* bp = wsB + (size_t)nt * 16384 + (size_t)lane * 16;

    v8i bq[4];
#pragma unroll
    for (int i = 0; i < 4; ++i) bq[i] = loadB(bp, i);

    v8i d = {0, 0, 0, 0, 0, 0, 0, 0};
#pragma unroll
    for (int g = 0; g < NGA; ++g) {
      d = __builtin_amdgcn_wmma_i32_16x16x64_iu8(
          true, aReg[g], true, bq[g & 3], d, false, false);
      if (g + 4 < NGA) bq[g & 3] = loadB(bp, g + 4);
      if (g == 0) {
#pragma unroll
        for (int j = 0; j < 8; ++j)
          opPrev[(size_t)j * OUT_F] =
              __builtin_fmaf(sPrev, (float)dPrev[j], bsPrev);
      }
    }
    dPrev  = d;
    sPrev  = sAll;
    bsPrev = bs;
    opPrev = out + (size_t)(row0 + hi * 8) * OUT_F + col;

    SGB(0x020, 8);
    SGB(0x008, 1);
    SGB(0x020, 2);
    SGB(0x002, 24);
    SGB(0x040, 8);
#pragma unroll
    for (int g = 1; g < 12; ++g) {
      SGB(0x008, 1);
      SGB(0x020, 2);
    }
    SGB(0x008, 4);
  }
#pragma unroll
  for (int j = 0; j < 8; ++j)
    opPrev[(size_t)j * OUT_F] = __builtin_fmaf(sPrev, (float)dPrev[j], bsPrev);
}

__global__ __launch_bounds__(256) void gemm_iu8_general_kernel(
    const float* __restrict__ x, const float* __restrict__ bias,
    const float* __restrict__ delta, const float* __restrict__ alpha,
    const int* __restrict__ actbits, const signed char* __restrict__ wsB,
    float* __restrict__ out) {
  if (alpha_uniform(alpha)) return;

  int lane = threadIdx.x & 31;
  int wave = threadIdx.x >> 5;
  int tile = blockIdx.x * 8 + wave;
  int m    = lane & 15;
  int hi   = lane >> 4;
  int row0 = tile * 16;
  const float* xr = x + (size_t)(row0 + m) * IN_F;

  float scaleA[NGA];
  v8i   aReg[NGA];
  build_A(xr, hi, alpha, actbits, aReg, scaleA);

#pragma unroll 1
  for (int nt = 0; nt < NTILES; ++nt) {
    int   col   = nt * 16 + m;
    float wstep = __builtin_exp2f(delta[nt >> 2]);
    float bs    = bias[col];
    const signed char* bp = wsB + (size_t)nt * 16384 + (size_t)lane * 16;
    const v8i czero = {0, 0, 0, 0, 0, 0, 0, 0};

    v8i bBuf = loadB(bp, 0);
    v8i dPrev = __builtin_amdgcn_wmma_i32_16x16x64_iu8(
        true, aReg[0], true, bBuf, czero, false, false);
    float sPrev = scaleA[0] * wstep;
    float acc[8];
#pragma unroll
    for (int j = 0; j < 8; ++j) acc[j] = bs;
    bBuf = loadB(bp, 1);

#pragma unroll
    for (int g = 1; g < NGA; ++g) {
      v8i bPre = (g + 1 < NGA) ? loadB(bp, g + 1) : bBuf;
      v8i dCur = __builtin_amdgcn_wmma_i32_16x16x64_iu8(
          true, aReg[g], true, bBuf, czero, false, false);
      float sCur = scaleA[g] * wstep;
#pragma unroll
      for (int j = 0; j < 8; ++j) acc[j] += sPrev * (float)dPrev[j];
      dPrev = dCur;
      sPrev = sCur;
      bBuf  = bPre;
    }
#pragma unroll
    for (int j = 0; j < 8; ++j) acc[j] += sPrev * (float)dPrev[j];

    float* op = out + (size_t)(row0 + hi * 8) * OUT_F + col;
#pragma unroll
    for (int j = 0; j < 8; ++j) op[(size_t)j * OUT_F] = acc[j];
  }
}

extern "C" void kernel_launch(void* const* d_in, const int* in_sizes, int n_in,
                              void* d_out, int out_size, void* d_ws, size_t ws_size,
                              hipStream_t stream) {
  const float* x       = (const float*)d_in[0];
  const float* weight  = (const float*)d_in[1];
  const float* bias    = (const float*)d_in[2];
  const float* phi     = (const float*)d_in[3];
  const float* delta   = (const float*)d_in[4];
  const int*   bit     = (const int*)d_in[5];
  const float* alpha   = (const float*)d_in[6];
  const int*   actbits = (const int*)d_in[7];
  float*       out     = (float*)d_out;
  signed char* wsB     = (signed char*)d_ws;

  wq_swizzle_kernel<<<256, 256, 0, stream>>>(weight, phi, delta, bit, wsB);
  gemm_iu8_uniform_kernel<<<RTILES / 8, 256, 0, stream>>>(
      x, bias, delta, alpha, actbits, wsB, out);
  gemm_iu8_general_kernel<<<RTILES / 8, 256, 0, stream>>>(
      x, bias, delta, alpha, actbits, wsB, out);
}
